// PicoDetHead_77403900609022
// MI455X (gfx1250) — matrix operand order repaired and checked by reading, not compiled, not run on hardware
//
#include <hip/hip_runtime.h>
#include <hip/hip_bf16.h>

typedef __attribute__((ext_vector_type(16))) _Float16 v16h;
typedef __attribute__((ext_vector_type(8)))  float    v8f;
typedef __attribute__((ext_vector_type(4)))  int      i4v;
typedef _Float16 f16;

#define CIN 96
#define LDM 72                 // K-major A-tile stride (f16): 64 pixels + pad, 16B-aligned rows
#define LDW (CIN + 8)          // Ws row stride (f16)
#define NCLS 80
#define NREG 36
#define NANCH 8400

#if __has_builtin(__builtin_amdgcn_global_load_async_to_lds_b128) && \
    __has_builtin(__builtin_amdgcn_s_wait_asynccnt)
#define HAVE_ASYNC 1
#else
#define HAVE_ASYNC 0
#endif

#if HAVE_ASYNC
typedef __attribute__((address_space(1))) i4v gi4v;   // global int4
typedef __attribute__((address_space(3))) i4v li4v;   // LDS int4
#endif

// ---------------------------------------------------------------------------
// Fused conv-as-GEMM with WMMA f32<=f16 16x16x32.
//   D[m,n] = sum_k A[m,k]*B[k,n]; A = activations (pixels x Cin, K-major in
//   LDS), B = W^T.  3x3 conv = 9 shifted 1x1 GEMMs into the same accumulators.
// Fast path (1x1, f16 src, no gate, full tile): activations staged straight
// into LDS with GLOBAL_LOAD_ASYNC_TO_LDS_B128 (ASYNCcnt), 16B per op.
// Epilogue: PRED ? (+bias -> f32 logits) : (BN scale/shift + ReLU6 -> f16),
// vectorized 8-pixel stores per lane.  All WMMA in uniform control flow.
// ---------------------------------------------------------------------------
template<int NT, int KS, bool SRCF32, bool PRED>
__global__ __launch_bounds__(128)
void conv_wmma(const void* __restrict__ src, const float* __restrict__ w,
               const float* __restrict__ e0,      // PRED: bias   | ghost: bn scale
               const float* __restrict__ e1,      // PRED: unused | ghost: bn shift
               const float* __restrict__ gate,    // [B][CIN] or null
               void* __restrict__ dst, int Cout, int H, int W)
{
  const int HW  = H * W;
  const int b   = blockIdx.y;
  const int p0  = blockIdx.x * 64;
  const int tid = threadIdx.x;
  const int lane = tid & 31;
  const int wv   = tid >> 5;        // wave id 0..3
  const int half = lane >> 4;       // hi/lo lane group
  const int l16  = lane & 15;

  __shared__ f16 As[CIN * LDM];     // K-major: As[k*LDM + m]
  __shared__ f16 Ws[NT * 16 * LDW]; // row-major: Ws[n*LDW + k]

  const float* sf = (const float*)src;
  const f16*   sh = (const f16*)src;

  // per-thread constants for the scalar staging path
  const int sm = tid & 63;          // pixel within tile
  const int cb = tid >> 6;          // channel phase (0/1)
  const int sp = p0 + sm;
  const int sh_ = (sp < HW) ? sp / W : 0;
  const int sw_ = (sp < HW) ? sp % W : 0;

  v8f acc[NT];
#pragma unroll
  for (int i = 0; i < NT; ++i) { v8f z = {}; acc[i] = z; }

#pragma unroll
  for (int t = 0; t < KS * KS; ++t) {
    const int dy = t / KS - KS / 2;
    const int dx = t % KS - KS / 2;

    // ---- stage weight tap slice: Ws[n][k], zero-pad n >= Cout -------------
    if constexpr (KS == 1) {
      for (int i = tid; i < NT * 16 * (CIN / 4); i += 128) {
        int co = i / (CIN / 4), c4 = (i % (CIN / 4)) * 4;
        float4 v4 = make_float4(0.f, 0.f, 0.f, 0.f);
        if (co < Cout) v4 = *(const float4*)&w[co * CIN + c4];
        union { f16 h[4]; uint2 q; } u;
        u.h[0] = (f16)v4.x; u.h[1] = (f16)v4.y; u.h[2] = (f16)v4.z; u.h[3] = (f16)v4.w;
        *(uint2*)&Ws[co * LDW + c4] = u.q;
      }
    } else {
      for (int i = tid; i < NT * 16 * CIN; i += 128) {
        int co = i / CIN, ci = i % CIN;
        float v = (co < Cout) ? w[((co * CIN + ci) * KS + (t / KS)) * KS + (t % KS)] : 0.f;
        Ws[co * LDW + ci] = (f16)v;
      }
    }

    // ---- stage activation tile (K-major) ----------------------------------
    const bool fast = (KS == 1) && !SRCF32 && (gate == nullptr) && (p0 + 64 <= HW);
    if (fast) {
      const f16* sbase = sh + (long)b * CIN * HW + p0;
#pragma unroll
      for (int r = 0; r < 6; ++r) {                 // 96ch x 64px x 2B / 16B / 128thr
        int q  = tid + 128 * r;
        int ci = q >> 3;
        int mb = (q & 7) << 3;                      // 8-pixel chunk
#if HAVE_ASYNC
        __builtin_amdgcn_global_load_async_to_lds_b128(
            (gi4v*)(f16*)(sbase + (long)ci * HW + mb),
            (li4v*)&As[ci * LDM + mb], 0, 0);
#else
        *(uint4*)&As[ci * LDM + mb] = *(const uint4*)(sbase + (long)ci * HW + mb);
#endif
      }
#if HAVE_ASYNC
      __builtin_amdgcn_s_wait_asynccnt(0);
#endif
    } else {
      const int hh = sh_ + dy, ww = sw_ + dx;
      const bool ok = (sp < HW) && hh >= 0 && hh < H && ww >= 0 && ww < W;
      const long poff = (long)hh * W + ww;
      for (int ci = cb; ci < CIN; ci += 2) {
        float v = 0.f;
        if (ok) {
          long idx = ((long)(b * CIN + ci)) * HW + poff;
          if constexpr (SRCF32) v = sf[idx]; else v = (float)sh[idx];
          if (gate) v *= gate[b * CIN + ci];
        }
        As[ci * LDM + sm] = (f16)v;
      }
    }
    __syncthreads();

    // ---- A fragments per K-chunk (ISA 16-bit A 16x32 layout, K-major LDS) -
    v16h afr[3];
    const int am = wv * 16 + l16;
#pragma unroll
    for (int c0 = 0; c0 < 3; ++c0) {
#pragma unroll
      for (int j = 0; j < 8; ++j) {
        int kk = 32 * c0 + ((j < 4) ? (half * 8 + 2 * j) : (16 + half * 8 + 2 * (j - 4)));
        afr[c0][2 * j]     = As[kk * LDM + am];
        afr[c0][2 * j + 1] = As[(kk + 1) * LDM + am];
      }
    }
    // ---- B fragments (two contiguous 16B LDS reads) + WMMA ----------------
#pragma unroll
    for (int nt = 0; nt < NT; ++nt) {
#pragma unroll
      for (int c0 = 0; c0 < 3; ++c0) {
        union { v16h v; uint4 q[2]; } bu;
        const f16* wrow = &Ws[(nt * 16 + l16) * LDW + 32 * c0 + half * 8];
        bu.q[0] = *(const uint4*)wrow;
        bu.q[1] = *(const uint4*)(wrow + 16);
        acc[nt] = __builtin_amdgcn_wmma_f32_16x16x32_f16(
            false, afr[c0], false, bu.v, (short)0, acc[nt], false, false);
      }
    }
    __syncthreads();
  }

  // ---- epilogue: lane owns 8 consecutive pixels (m = 8*half + j) ----------
  const int pbase = p0 + wv * 16 + half * 8;
  const bool vfull = (p0 + 64 <= HW);
#pragma unroll
  for (int nt = 0; nt < NT; ++nt) {
    int n = nt * 16 + l16;
    if (n >= Cout) continue;
    if constexpr (PRED) {
      float* dp = (float*)dst + ((long)(b * Cout + n)) * HW + pbase;
      if (vfull) {
        float4 q0, q1;
        q0.x = acc[nt][0] + e0[n]; q0.y = acc[nt][1] + e0[n];
        q0.z = acc[nt][2] + e0[n]; q0.w = acc[nt][3] + e0[n];
        q1.x = acc[nt][4] + e0[n]; q1.y = acc[nt][5] + e0[n];
        q1.z = acc[nt][6] + e0[n]; q1.w = acc[nt][7] + e0[n];
        *(float4*)dp = q0; *(float4*)(dp + 4) = q1;
      } else {
#pragma unroll
        for (int j = 0; j < 8; ++j)
          if (pbase + j < HW) dp[j] = acc[nt][j] + e0[n];
      }
    } else {
      f16* dp = (f16*)dst + ((long)(b * CIN + n)) * HW + pbase;
      if (vfull) {
        union { f16 h[8]; uint4 q; } u;
#pragma unroll
        for (int j = 0; j < 8; ++j) {
          float v = acc[nt][j] * e0[n] + e1[n];
          u.h[j] = (f16)fminf(fmaxf(v, 0.f), 6.f);
        }
        *(uint4*)dp = u.q;
      } else {
#pragma unroll
        for (int j = 0; j < 8; ++j)
          if (pbase + j < HW) {
            float v = acc[nt][j] * e0[n] + e1[n];
            dp[j] = (f16)fminf(fmaxf(v, 0.f), 6.f);
          }
      }
    }
  }
}

// ---------------------------------------------------------------------------
// GhostConv cheap branch: depthwise 3x3 on channels [0,48) -> [48,96) + BN + ReLU6
// ---------------------------------------------------------------------------
__global__ void ghost_dw(f16* __restrict__ buf, const float* __restrict__ dw,
                         const float* __restrict__ s, const float* __restrict__ bb,
                         int H, int W)
{
  int HW = H * W;
  int p = blockIdx.x * 256 + threadIdx.x;
  int c = blockIdx.y;   // 0..47
  int b = blockIdx.z;
  if (p >= HW) return;
  int h = p / W, ww0 = p % W;
  const f16* y = buf + ((long)(b * CIN + c)) * HW;
  float a = 0.f;
#pragma unroll
  for (int ty = 0; ty < 3; ++ty)
#pragma unroll
    for (int tx = 0; tx < 3; ++tx) {
      int hh = h + ty - 1, ww = ww0 + tx - 1;
      if (hh >= 0 && hh < H && ww >= 0 && ww < W)
        a += (float)y[hh * W + ww] * dw[c * 9 + ty * 3 + tx];
    }
  float v = fminf(fmaxf(a * s[c] + bb[c], 0.f), 6.f);
  buf[((long)(b * CIN + 48 + c)) * HW + p] = (f16)v;
}

// ---------------------------------------------------------------------------
// ESE: global average pool, then gate = sigmoid(pooled @ W^T)
// ---------------------------------------------------------------------------
__global__ void pool_mean(const f16* __restrict__ buf, float* __restrict__ pooled, int HW)
{
  int c = blockIdx.x, b = blockIdx.y;
  const f16* x = buf + ((long)(b * CIN + c)) * HW;
  float s = 0.f;
  for (int i = threadIdx.x; i < HW; i += 256) s += (float)x[i];
  __shared__ float red[256];
  red[threadIdx.x] = s; __syncthreads();
  for (int st = 128; st > 0; st >>= 1) {
    if (threadIdx.x < st) red[threadIdx.x] += red[threadIdx.x + st];
    __syncthreads();
  }
  if (threadIdx.x == 0) pooled[b * CIN + c] = red[0] / (float)HW;
}

__global__ void ese_gate(const float* __restrict__ pooled, const float* __restrict__ wese,
                         float* __restrict__ gate)
{
  int i = blockIdx.x * 256 + threadIdx.x;   // 16*96
  if (i >= 16 * CIN) return;
  int b = i / CIN, o = i % CIN;
  float s = 0.f;
  for (int c = 0; c < CIN; ++c) s += pooled[b * CIN + c] * wese[o * CIN + c];
  gate[i] = 1.f / (1.f + __expf(-s));
}

// ---------------------------------------------------------------------------
// Decode: sigmoid scores, per-anchor max, per-(b,level) atomic max, DFL boxes
// ---------------------------------------------------------------------------
__global__ void decode_kernel(const float* __restrict__ clsl, const float* __restrict__ regl,
                              float* __restrict__ boxes, float* __restrict__ scores,
                              float* __restrict__ maxsc, int* __restrict__ lvmax,
                              int H, int W, int anchor0, float stride, int lvl)
{
  int HW = H * W;
  int p = blockIdx.x * 128 + threadIdx.x;
  int b = blockIdx.y;
  if (p >= HW) return;
  long a = (long)b * NANCH + anchor0 + p;
  float mx = 0.f;
  for (int k = 0; k < NCLS; ++k) {
    float lg = clsl[((long)(b * NCLS + k)) * HW + p];
    float sc = 1.f / (1.f + __expf(-lg));
    scores[a * NCLS + k] = sc;
    mx = fmaxf(mx, sc);
  }
  maxsc[a] = mx;
  atomicMax(&lvmax[lvl * 16 + b], __float_as_int(mx));   // scores > 0 -> int-ordered
  float ax = ((float)(p % W) + 0.5f) * stride;
  float ay = ((float)(p / W) + 0.5f) * stride;
  float d[4];
#pragma unroll
  for (int s4 = 0; s4 < 4; ++s4) {
    float v[9], m2 = -1e30f;
#pragma unroll
    for (int j = 0; j < 9; ++j) {
      v[j] = regl[((long)(b * NREG + s4 * 9 + j)) * HW + p];
      m2 = fmaxf(m2, v[j]);
    }
    float den = 0.f, num = 0.f;
#pragma unroll
    for (int j = 0; j < 9; ++j) {
      float e = __expf(v[j] - m2);
      den += e; num += e * (float)j;
    }
    d[s4] = (num / den) * stride;
  }
  boxes[a * 4 + 0] = ax - d[0];
  boxes[a * 4 + 1] = ay - d[1];
  boxes[a * 4 + 2] = ax + d[2];
  boxes[a * 4 + 3] = ay + d[3];
}

__global__ void mask_kernel(float* __restrict__ boxes, float* __restrict__ scores,
                            const float* __restrict__ maxsc, const int* __restrict__ lvmax)
{
  long i = (long)blockIdx.x * 256 + threadIdx.x;
  if (i >= 16L * NANCH) return;
  int aa = (int)(i % NANCH);
  int b  = (int)(i / NANCH);
  int lvl = aa < 6400 ? 0 : (aa < 8000 ? 1 : 2);
  float ms = maxsc[i];
  float gm = __int_as_float(lvmax[lvl * 16 + b]);
  bool keep = (ms > 0.01f) || (ms == gm);
  if (!keep) {
    for (int k = 0; k < NCLS; ++k) scores[i * NCLS + k] = 0.f;
    for (int k = 0; k < 4; ++k)    boxes[i * 4 + k]     = 0.f;
  }
}

__global__ void init_lvmax(int* lvmax)
{
  if (threadIdx.x < 64) lvmax[threadIdx.x] = 0;
}

// ---------------------------------------------------------------------------
// Host orchestration.
// Input order (jax pytree, sorted dict keys): feat0, feat1, feat2, then per
// level (35 leaves): cls_conv[0..2] x {bn1_b,bn1_s,bn2_b,bn2_s,dw_w,pw_w},
// cls_pred_b, cls_pred_w, ese_w, reg_conv[0..1] x {...}, reg_pred_b, reg_pred_w.
// Output: boxes [16,8400,4] then scores [16,8400,80], flat f32.
// ---------------------------------------------------------------------------
extern "C" void kernel_launch(void* const* d_in, const int* in_sizes, int n_in,
                              void* d_out, int out_size, void* d_ws, size_t ws_size,
                              hipStream_t stream)
{
  (void)in_sizes; (void)n_in; (void)out_size; (void)ws_size;
  const int   Hs[3]      = {80, 40, 20};
  const float strides[3] = {8.f, 16.f, 32.f};
  const int   anch0[3]   = {0, 6400, 8000};

  // workspace carving (all chunks 256B-aligned by construction)
  char* ws = (char*)d_ws;
  const size_t BUFB = (size_t)16 * CIN * 6400 * sizeof(f16);      // 19.66 MB
  f16* bufA = (f16*)ws; ws += BUFB;
  f16* bufB = (f16*)ws; ws += BUFB;
  f16* bufC = (f16*)ws; ws += BUFB;
  float* clsl   = (float*)ws; ws += (size_t)16 * NCLS * 6400 * sizeof(float);
  float* regl   = (float*)ws; ws += (size_t)16 * NREG * 6400 * sizeof(float);
  float* pooled = (float*)ws; ws += 16 * CIN * sizeof(float);
  float* gate   = (float*)ws; ws += 16 * CIN * sizeof(float);
  float* maxsc  = (float*)ws; ws += (size_t)16 * NANCH * sizeof(float);
  int*   lvmax  = (int*)ws;   ws += 256;

  float* boxes  = (float*)d_out;
  float* scores = boxes + (size_t)16 * NANCH * 4;

  init_lvmax<<<dim3(1), dim3(64), 0, stream>>>(lvmax);

  for (int l = 0; l < 3; ++l) {
    const int H = Hs[l], W = Hs[l], HW = H * W;
    const float* feat = (const float*)d_in[l];
    const int base = 3 + l * 35;
    auto P = [&](int k) { return (const float*)d_in[base + k]; };
    // ghost leaf offsets: +0 bn1_b, +1 bn1_s, +2 bn2_b, +3 bn2_s, +4 dw_w, +5 pw_w
    // 18 cls_pred_b, 19 cls_pred_w, 20 ese_w, 33 reg_pred_b, 34 reg_pred_w

    dim3 cgrid((HW + 63) / 64, 16);
    dim3 dgrid((HW + 255) / 256, 48, 16);

    // cls ghost 0 (3x3 pw): feat -> bufA[0:48], dw -> bufA[48:96]
    conv_wmma<3, 3, true, false><<<cgrid, 128, 0, stream>>>(
        feat, P(5), P(1), P(0), nullptr, bufA, 48, H, W);
    ghost_dw<<<dgrid, 256, 0, stream>>>(bufA, P(4), P(3), P(2), H, W);
    // cls ghost 1: bufA -> bufB
    conv_wmma<3, 1, false, false><<<cgrid, 128, 0, stream>>>(
        bufA, P(6 + 5), P(6 + 1), P(6 + 0), nullptr, bufB, 48, H, W);
    ghost_dw<<<dgrid, 256, 0, stream>>>(bufB, P(6 + 4), P(6 + 3), P(6 + 2), H, W);
    // cls ghost 2: bufB -> bufA
    conv_wmma<3, 1, false, false><<<cgrid, 128, 0, stream>>>(
        bufB, P(12 + 5), P(12 + 1), P(12 + 0), nullptr, bufA, 48, H, W);
    ghost_dw<<<dgrid, 256, 0, stream>>>(bufA, P(12 + 4), P(12 + 3), P(12 + 2), H, W);

    // ESE gate from pre-gate cls features
    pool_mean<<<dim3(96, 16), 256, 0, stream>>>(bufA, pooled, HW);
    ese_gate<<<dim3(6), 256, 0, stream>>>(pooled, P(20), gate);

    // reg ghost 0: feat -> bufB ; reg ghost 1: bufB -> bufC
    conv_wmma<3, 1, true, false><<<cgrid, 128, 0, stream>>>(
        feat, P(21 + 5), P(21 + 1), P(21 + 0), nullptr, bufB, 48, H, W);
    ghost_dw<<<dgrid, 256, 0, stream>>>(bufB, P(21 + 4), P(21 + 3), P(21 + 2), H, W);
    conv_wmma<3, 1, false, false><<<cgrid, 128, 0, stream>>>(
        bufB, P(27 + 5), P(27 + 1), P(27 + 0), nullptr, bufC, 48, H, W);
    ghost_dw<<<dgrid, 256, 0, stream>>>(bufC, P(27 + 4), P(27 + 3), P(27 + 2), H, W);

    // preds: cls (gate folded into A-tile load), reg
    conv_wmma<5, 1, false, true><<<cgrid, 128, 0, stream>>>(
        bufA, P(19), P(18), nullptr, gate, clsl, NCLS, H, W);
    conv_wmma<3, 1, false, true><<<cgrid, 128, 0, stream>>>(
        bufC, P(34), P(33), nullptr, nullptr, regl, NREG, H, W);

    decode_kernel<<<dim3((HW + 127) / 128, 16), 128, 0, stream>>>(
        clsl, regl, boxes, scores, maxsc, lvmax, H, W, anch0[l], strides[l], l);
  }

  mask_kernel<<<dim3((16 * NANCH + 255) / 256), 256, 0, stream>>>(boxes, scores, maxsc, lvmax);
}
